// DeepSpeechCell_15539191677540
// MI455X (gfx1250) — compile-verified, matrix operand order checked by reading, not run on hardware
//
#include <hip/hip_runtime.h>
#include <hip/hip_bf16.h>

// DeepSpeech cell: H = x[B,T,D] @ W[D,S] + b ; out_t = relu(H_t + out_{t-1} @ Wr[S,S])
#define BB 64
#define TT 512
#define DD 1024
#define SS 2048
#define NT 4            // N-tiles (16 cols) per wave: 4 independent WMMA acc chains
#define SCAN_BLOCKS 16  // 4 mTiles x 4 nGroups; all co-resident -> grid barrier is safe

typedef __bf16 bf16_t;
typedef bf16_t v16bf __attribute__((ext_vector_type(16)));
typedef bf16_t v8bf  __attribute__((ext_vector_type(8)));
typedef float  v8f   __attribute__((ext_vector_type(8)));
typedef float  v4f   __attribute__((ext_vector_type(4)));
typedef unsigned short v4us __attribute__((ext_vector_type(4)));

// fp32 -> bf16 bits, round-to-nearest-even.
__device__ __forceinline__ unsigned short f2bf_bits(float f) {
  union { float f; unsigned u; } in; in.f = f;
  unsigned r = in.u + 0x7FFFu + ((in.u >> 16) & 1u);
  return (unsigned short)(r >> 16);
}

// One 16-bit WMMA operand fragment: two 16B chunks into the two halves of the
// v16bf register block (ISA 7.12.2 layout; khalf selects the lane's K chunks).
__device__ __forceinline__ v16bf load_frag(const bf16_t* __restrict__ row,
                                           int k0, int k1) {
  v16bf f;
  *reinterpret_cast<v8bf*>(&f)       = *(const v8bf*)(row + k0);
  *(reinterpret_cast<v8bf*>(&f) + 1) = *(const v8bf*)(row + k1);
  return f;
}

template <int K>
__device__ __forceinline__ void load_set(const bf16_t* __restrict__ Arow,
                                         const bf16_t* __restrict__ Brow,
                                         int kbase, int khalf,
                                         v16bf& a, v16bf b[NT]) {
  const int k0 = kbase + khalf, k1 = k0 + 16;
  a = load_frag(Arow, k0, k1);
#pragma unroll
  for (int j = 0; j < NT; ++j)
    b[j] = load_frag(Brow + (size_t)j * 16 * K, k0, k1);
}

__device__ __forceinline__ void compute_set(const v16bf& a, const v16bf b[NT],
                                            v8f acc[NT]) {
#pragma unroll
  for (int j = 0; j < NT; ++j)
    acc[j] = __builtin_amdgcn_wmma_f32_16x16x32_bf16(
        false, a, false, b[j], (short)0, acc[j], false, false);
}

// 16 x (16*NT) tile GEMM, 1-deep software pipeline with two register sets
// (loads for step k+1 in flight while WMMAs consume step k).
template <int K>
__device__ __forceinline__ void wmma_tile_nt4(const bf16_t* __restrict__ Arow,
                                              const bf16_t* __restrict__ Brow,
                                              int khalf, v8f acc[NT]) {
  static_assert((K / 32) % 2 == 0, "even K-step count required");
  v16bf a0, a1;
  v16bf b0[NT], b1[NT];
  load_set<K>(Arow, Brow, 0, khalf, a0, b0);
  for (int kk = 32; kk < K - 32; kk += 64) {
    load_set<K>(Arow, Brow, kk, khalf, a1, b1);
    compute_set(a0, b0, acc);
    load_set<K>(Arow, Brow, kk + 32, khalf, a0, b0);
    compute_set(a1, b1, acc);
  }
  load_set<K>(Arow, Brow, K - 32, khalf, a1, b1);
  compute_set(a0, b0, acc);
  compute_set(a1, b1, acc);
}

// ---------------------------------------------------------------------------
// Prep kernels
// ---------------------------------------------------------------------------
// fp32 [K x N] row-major -> bf16 [N x K] row-major. K power-of-2 => shifts.
template <int K, int N>
__global__ void __launch_bounds__(256)
cvt_transpose_bf16(const float* __restrict__ src, unsigned short* __restrict__ dst) {
  size_t idx = (size_t)blockIdx.x * 256 + threadIdx.x;
  if (idx >= (size_t)K * N) return;
  int k = (int)(idx & (K - 1));
  int n = (int)(idx / K);
  dst[idx] = f2bf_bits(src[(size_t)k * N + n]);
}

// Straight fp32 -> bf16 convert, 4 elements/thread (b128 in, b64 out).
__global__ void __launch_bounds__(256)
cvt_bf16x4(const float* __restrict__ src, unsigned short* __restrict__ dst) {
  size_t i4 = ((size_t)blockIdx.x * 256 + threadIdx.x) * 4;
  v4f x = *(const v4f*)(src + i4);
  v4us o;
#pragma unroll
  for (int i = 0; i < 4; ++i) o[i] = f2bf_bits(x[i]);
  *(v4us*)(dst + i4) = o;
}

__global__ void __launch_bounds__(64)
init_barrier(unsigned int* __restrict__ bar) {
  if (threadIdx.x == 0) *bar = 0u;
}

// ---------------------------------------------------------------------------
// Stage 1: H[(b*T+t), n] = sum_k xbf[(b*T+t), k] * W[k, n] + bias[n]
// H is a 256 MB stream written once and read once -> non-temporal stores so
// it does not evict the L2-resident weights/state used by the scan.
// ---------------------------------------------------------------------------
__global__ void __launch_bounds__(256)
ds_input_gemm(const unsigned short* __restrict__ Xbf,
              const unsigned short* __restrict__ WT,
              const float* __restrict__ bias, float* __restrict__ H) {
  const int wave  = threadIdx.x >> 5;
  const int lane  = threadIdx.x & 31;
  const int r     = lane & 15;
  const int khalf = (lane >> 4) << 3;                 // 0 or 8
  const int mTile     = blockIdx.x >> 2;              // 2048 mTiles x 4 nGroups
  const int nTileBase = ((blockIdx.x & 3) * 8 + wave) * NT;

  const bf16_t* Arow = (const bf16_t*)Xbf + (size_t)(mTile * 16 + r) * DD;
  const bf16_t* Brow = (const bf16_t*)WT  + (size_t)(nTileBase * 16 + r) * DD;

  v8f acc[NT];
#pragma unroll
  for (int j = 0; j < NT; ++j) acc[j] = (v8f){0.f,0.f,0.f,0.f,0.f,0.f,0.f,0.f};

  wmma_tile_nt4<DD>(Arow, Brow, khalf, acc);

  const int rowbase = mTile * 16 + (lane >> 4) * 8;   // C layout: lanes 16-31 -> M+8
#pragma unroll
  for (int j = 0; j < NT; ++j) {
    const int n = (nTileBase + j) * 16 + r;
    const float bv = bias[n];
    float* Cp = H + (size_t)rowbase * SS + n;
#pragma unroll
    for (int i = 0; i < 8; ++i)
      __builtin_nontemporal_store(acc[j][i] + bv, Cp + (size_t)i * SS);
  }
}

// ---------------------------------------------------------------------------
// t = 0: out_0 = relu(H_0); seed the bf16 ping buffer for step 1.
// ---------------------------------------------------------------------------
__global__ void __launch_bounds__(256)
relu_t0(float* __restrict__ out, unsigned short* __restrict__ outbf) {
  int idx = blockIdx.x * 256 + threadIdx.x;           // < BB*SS
  int b = idx / SS, s = idx % SS;
  size_t off = (size_t)b * TT * SS + s;
  float v = __builtin_nontemporal_load(out + off);
  v = v > 0.f ? v : 0.f;
  __builtin_nontemporal_store(v, out + off);
  outbf[idx] = f2bf_bits(v);                          // [BB, SS] row-major
}

// ---------------------------------------------------------------------------
// Persistent scan kernel: loops t = 1..T-1 with a device-scope counting
// barrier between steps (monotonic target 16*t; 16 blocks are co-resident).
// Removes 510 kernel-launch boundaries from the latency-critical chain and
// keeps each wave's Wr^T tile base addresses live for the whole scan.
//   step t: out[b,t,n] = relu(H[b,t,n] + sum_k prev_bf[b,k] * Wr[k,n])
// prev/cur bf16 state ping-pongs in two L2-resident 256 KB buffers.
// ---------------------------------------------------------------------------
__global__ void __launch_bounds__(256)
ds_recurrent_scan(const unsigned short* __restrict__ WrT,
                  float* __restrict__ out,
                  unsigned short* __restrict__ pp0,
                  unsigned short* __restrict__ pp1,
                  unsigned int* __restrict__ bar) {
  const int wave  = threadIdx.x >> 5;
  const int lane  = threadIdx.x & 31;
  const int r     = lane & 15;
  const int khalf = (lane >> 4) << 3;
  const int mTile     = blockIdx.x >> 2;              // 0..3   (M = 64)
  const int nTileBase = ((blockIdx.x & 3) * 8 + wave) * NT;
  const size_t rowStride = (size_t)TT * SS;           // batch stride in d_out
  const int rowbase = mTile * 16 + (lane >> 4) * 8;

  const bf16_t* Brow = (const bf16_t*)WrT + (size_t)(nTileBase * 16 + r) * SS;

  for (int t = 1; t < TT; ++t) {
    const unsigned short* Abf = (t & 1) ? pp0 : pp1;
    unsigned short*       Cbf = (t & 1) ? pp1 : pp0;
    const bf16_t* Arow = (const bf16_t*)Abf + (size_t)(mTile * 16 + r) * SS;
    float* C = out + (size_t)t * SS;

    v8f acc[NT];
#pragma unroll
    for (int j = 0; j < NT; ++j) acc[j] = (v8f){0.f,0.f,0.f,0.f,0.f,0.f,0.f,0.f};

    wmma_tile_nt4<SS>(Arow, Brow, khalf, acc);

#pragma unroll
    for (int j = 0; j < NT; ++j) {
      const int n = (nTileBase + j) * 16 + r;
      float* Cp = C + (size_t)rowbase * rowStride + n;
      unsigned short* Bp = Cbf + (size_t)rowbase * SS + n;
#pragma unroll
      for (int i = 0; i < 8; ++i) {
        float v = __builtin_nontemporal_load(Cp + i * rowStride) + acc[j][i];
        v = v > 0.f ? v : 0.f;
        __builtin_nontemporal_store(v, Cp + i * rowStride);  // fp32 result
        Bp[(size_t)i * SS] = f2bf_bits(v);                   // bf16 A for t+1
      }
    }

    // ---- device-scope grid barrier (release stores, then count to 16*t) ----
    __threadfence();                 // make this step's stores device-visible
    __syncthreads();                 // whole block done with step t
    if (threadIdx.x == 0) {
      __hip_atomic_fetch_add(bar, 1u, __ATOMIC_RELEASE, __HIP_MEMORY_SCOPE_AGENT);
      const unsigned target = (unsigned)(SCAN_BLOCKS * t);
      while (__hip_atomic_load(bar, __ATOMIC_ACQUIRE, __HIP_MEMORY_SCOPE_AGENT) < target)
        __builtin_amdgcn_s_sleep(1);
    }
    __syncthreads();                 // release all waves into step t+1
    __threadfence();                 // acquire: no stale state reads
  }
}

// ---------------------------------------------------------------------------
extern "C" void kernel_launch(void* const* d_in, const int* in_sizes, int n_in,
                              void* d_out, int out_size, void* d_ws, size_t ws_size,
                              hipStream_t stream) {
  const float* x    = (const float*)d_in[0];   // [B,T,D]
  const float* W    = (const float*)d_in[1];   // [D,S]
  const float* Wr   = (const float*)d_in[2];   // [S,S]
  const float* bias = (const float*)d_in[3];   // [S]
  float* out = (float*)d_out;                  // [B,T,S]

  // Workspace (bf16 elements unless noted):
  //   WT  = W^T  [S x D]   4 MB     WrT = Wr^T [S x S]   8 MB
  //   Xbf = x    [B*T x D] 64 MB    pp0/pp1 [B x S]      256 KB each
  //   bar : one u32 grid-barrier counter
  unsigned short* WT  = (unsigned short*)d_ws;
  unsigned short* WrT = WT  + (size_t)SS * DD;
  unsigned short* Xbf = WrT + (size_t)SS * SS;
  unsigned short* pp0 = Xbf + (size_t)BB * TT * DD;
  unsigned short* pp1 = pp0 + (size_t)BB * SS;
  unsigned int*   bar = (unsigned int*)(pp1 + (size_t)BB * SS);

  // 1) Weight transpose+convert, activation convert, barrier reset.
  {
    size_t total = (size_t)DD * SS;
    cvt_transpose_bf16<DD, SS><<<(unsigned)((total + 255) / 256), 256, 0, stream>>>(W, WT);
  }
  {
    size_t total = (size_t)SS * SS;
    cvt_transpose_bf16<SS, SS><<<(unsigned)((total + 255) / 256), 256, 0, stream>>>(Wr, WrT);
  }
  {
    size_t total4 = (size_t)BB * TT * DD / 4;
    cvt_bf16x4<<<(unsigned)((total4 + 255) / 256), 256, 0, stream>>>(x, Xbf);
  }
  init_barrier<<<1, 64, 0, stream>>>(bar);

  // 2) Input-projection GEMM: H (+bias) straight into d_out.
  {
    int blocks = ((BB * TT) / 16) * (SS / (16 * NT * 8));  // 2048 * 4 = 8192
    ds_input_gemm<<<blocks, 256, 0, stream>>>(Xbf, WT, bias, out);
  }

  // 3) Sequential scan: relu at t=0, then ONE persistent kernel for t=1..511.
  relu_t0<<<(BB * SS) / 256, 256, 0, stream>>>(out, pp0);
  ds_recurrent_scan<<<SCAN_BLOCKS, 256, 0, stream>>>(WrT, out, pp0, pp1, bar);
}
